// VideoModel_23819888624008
// MI455X (gfx1250) — matrix operand order repaired and checked by reading, not compiled, not run on hardware
//
#include <hip/hip_runtime.h>

// ---------------------------------------------------------------------------
// Types for CDNA5 WMMA (wave32, 16x16x32 bf16 -> f32)
// ---------------------------------------------------------------------------
typedef __attribute__((ext_vector_type(16))) __bf16 v16bf;
typedef __attribute__((ext_vector_type(8)))  float  v8f;

static constexpr int Bb = 64;     // batch
static constexpr int Tt = 128;    // timesteps
static constexpr int Dd = 4096;   // input dim
static constexpr int Pp = 2048;   // projection dim
static constexpr int Hh = 512;    // hidden
static constexpr int Gg = 4 * Hh; // 2048 gate dim
static constexpr int Cc = 22;     // classes
static constexpr int Mrows = Bb * Tt; // 8192

// ---------------------------------------------------------------------------
// Helpers
// ---------------------------------------------------------------------------
__device__ __forceinline__ unsigned short f32_to_bf16(float f) {
    unsigned int u = __builtin_bit_cast(unsigned int, f);
    unsigned int r = (u + 0x7FFFu + ((u >> 16) & 1u)) >> 16; // RNE
    return (unsigned short)r;
}

// A fragment: 16x32 bf16 tile, row-major source, lda in elements.
// ISA layout: lanes 0-15 hold row M=lane, K {0..7,16..23}; lanes 16-31 hold
// row M=lane-16, K {8..15,24..31}. Two 16B loads per lane.
__device__ __forceinline__ v16bf load_frag_a(const unsigned short* base, int lda) {
    int lane = threadIdx.x & 31;
    int row  = lane & 15;
    int kh   = (lane >> 4) ? 8 : 0;
    const unsigned short* p = base + row * lda + kh;
    union { uint4 u[2]; v16bf v; } f;
    f.u[0] = *(const uint4*)(p);
    f.u[1] = *(const uint4*)(p + 16);
    return f.v;
}

// B fragment: 32x16 bf16 tile, where B = W^T and W is [N,K] row-major.
// Column n of B == row n of W (contiguous K). Lanes 0-15: N=lane, K {0..7,16..23};
// lanes 16-31: N=lane-16, K {8..15,24..31}. Two 16B loads per lane.
__device__ __forceinline__ v16bf load_frag_b(const unsigned short* wbase, int ldk) {
    int lane = threadIdx.x & 31;
    int col  = lane & 15;
    int koff = (lane >> 4) ? 8 : 0;
    const unsigned short* p = wbase + col * ldk + koff;
    union { uint4 u[2]; v16bf v; } f;
    f.u[0] = *(const uint4*)(p);
    f.u[1] = *(const uint4*)(p + 16);
    return f.v;
}

// C/D layout: VGPR i -> row (i + 8*(lane>=16)), col (lane&15).
__device__ __forceinline__ int d_row(int i) { return i + (((threadIdx.x & 31) >> 4) << 3); }
__device__ __forceinline__ int d_col()      { return (threadIdx.x & 31) & 15; }

__device__ __forceinline__ float sigmoidf(float x) { return 1.0f / (1.0f + __expf(-x)); }

// ---------------------------------------------------------------------------
// Core 64x64-per-wave MMA tile: 4 M-tiles x 4 N-tiles, 16 f32 accumulators.
// Per K-step: 16 independent b128 loads -> 16 independent WMMAs, so the
// scheduler can clause the loads and burst the MMAs without draining loadcnt
// before each one. Prefetch (global_prefetch_b8) runs 2 K-steps ahead.
// ---------------------------------------------------------------------------
__device__ __forceinline__ void mma_tile_4x4(const unsigned short* __restrict__ A,
                                             const unsigned short* __restrict__ W,
                                             int m0, int n0, int K,
                                             v8f acc[4][4]) {
    int lane16 = (threadIdx.x & 31) & 15;
    for (int k0 = 0; k0 < K; k0 += 32) {
        if (k0 + 64 < K) {
            // one byte/lane pulls the whole next-next cacheline
            __builtin_prefetch(A + (size_t)(m0 + lane16) * K + k0 + 64, 0, 1);
            __builtin_prefetch(W + (size_t)(n0 + lane16) * K + k0 + 64, 0, 1);
        }
        v16bf a[4], b[4];
#pragma unroll
        for (int i = 0; i < 4; ++i)
            a[i] = load_frag_a(A + (size_t)(m0 + 16 * i) * K + k0, K);
#pragma unroll
        for (int j = 0; j < 4; ++j)
            b[j] = load_frag_b(W + (size_t)(n0 + 16 * j) * K + k0, K);
#pragma unroll
        for (int i = 0; i < 4; ++i)
#pragma unroll
            for (int j = 0; j < 4; ++j)
                acc[i][j] = __builtin_amdgcn_wmma_f32_16x16x32_bf16(
                    false, a[i], false, b[j], (short)0, acc[i][j], false, false);
    }
}

// ---------------------------------------------------------------------------
// fp32 -> bf16 conversion
// ---------------------------------------------------------------------------
__global__ void cvt_f32_bf16(const float* __restrict__ in,
                             unsigned short* __restrict__ out, int n) {
    int i = blockIdx.x * blockDim.x + threadIdx.x;
    if (i < n) out[i] = f32_to_bf16(in[i]);
}

// ---------------------------------------------------------------------------
// Zero-init LSTM state (c fp32, h0 bf16)
// ---------------------------------------------------------------------------
__global__ void init_state(float* __restrict__ c, unsigned short* __restrict__ h0, int n) {
    int i = blockIdx.x * blockDim.x + threadIdx.x;
    if (i < n) { c[i] = 0.0f; h0[i] = 0; }
}

// ---------------------------------------------------------------------------
// GEMM1: lin = relu(x @ W1^T + b1)  [8192,4096] x [2048,4096]^T -> bf16 [8192,2048]
// 64x64 per wave: (8192/64)*(2048/64) = 4096 waves = 512 blocks of 8 waves.
// ---------------------------------------------------------------------------
__global__ void __launch_bounds__(256)
gemm1_relu(const unsigned short* __restrict__ A,   // x bf16 [M,K]
           const unsigned short* __restrict__ W,   // W1 bf16 [N,K]
           const float* __restrict__ bias,
           unsigned short* __restrict__ out) {     // lin bf16 [M,N]
    constexpr int K = Dd, N = Pp;
    int wave = blockIdx.x * 8 + (threadIdx.x >> 5);
    int m0 = (wave >> 5) * 64;   // 128 M blocks
    int n0 = (wave & 31) * 64;   // 32 N blocks

    v8f acc[4][4] = {};
    mma_tile_4x4(A, W, m0, n0, K, acc);

    int col0 = d_col();
#pragma unroll
    for (int j = 0; j < 4; ++j) {
        int col = n0 + 16 * j + col0;
        float bv = bias[col];
#pragma unroll
        for (int i = 0; i < 4; ++i) {
#pragma unroll
            for (int r = 0; r < 8; ++r) {
                int row = m0 + 16 * i + d_row(r);
                float v = acc[i][j][r] + bv;
                v = v > 0.0f ? v : 0.0f;
                out[(size_t)row * N + col] = f32_to_bf16(v);
            }
        }
    }
}

// ---------------------------------------------------------------------------
// GEMM2: xg[t,b,:] = lin[b,t,:] @ W_ih^T + b_ih + b_hh  -> fp32 [T,B,4H]
// ---------------------------------------------------------------------------
__global__ void __launch_bounds__(256)
gemm2_xg(const unsigned short* __restrict__ A,   // lin bf16 [M=B*T, K=P]
         const unsigned short* __restrict__ W,   // W_ih bf16 [4H, P]
         const float* __restrict__ b_ih,
         const float* __restrict__ b_hh,
         float* __restrict__ xg) {               // [T,B,4H]
    constexpr int K = Pp, N = Gg;
    int wave = blockIdx.x * 8 + (threadIdx.x >> 5);
    int m0 = (wave >> 5) * 64;
    int n0 = (wave & 31) * 64;

    v8f acc[4][4] = {};
    mma_tile_4x4(A, W, m0, n0, K, acc);

    int col0 = d_col();
#pragma unroll
    for (int j = 0; j < 4; ++j) {
        int col = n0 + 16 * j + col0;
        float bv = b_ih[col] + b_hh[col];
#pragma unroll
        for (int i = 0; i < 4; ++i) {
#pragma unroll
            for (int r = 0; r < 8; ++r) {
                int row = m0 + 16 * i + d_row(r);  // row = b*T + t
                int b = row >> 7;                  // /T (T=128)
                int t = row & (Tt - 1);
                xg[((size_t)t * Bb + b) * N + col] = acc[i][j][r] + bv;
            }
        }
    }
}

// ---------------------------------------------------------------------------
// Fused LSTM step. Grid: 16 blocks x 256 thr = 128 waves == 4 B-tiles x 32 H-tiles.
// Each wave computes all four gate tiles for its (b,h) 16x16 patch. B-fragment
// loads are hoisted into an array so each K-step is one load clause + one
// 4-WMMA burst (latency-critical: 128 serial steps).
// ---------------------------------------------------------------------------
__global__ void __launch_bounds__(256)
lstm_step(const unsigned short* __restrict__ h_prev,  // bf16 [B,H]
          const unsigned short* __restrict__ Whh,     // bf16 [4H,H]
          const float* __restrict__ xg_t,             // fp32 [B,4H]
          float* __restrict__ c_state,                // fp32 [B,H]
          unsigned short* __restrict__ h_next,        // bf16 [B,H]
          float* __restrict__ hs_t) {                 // fp32 [B,H]
    int wave = blockIdx.x * 8 + (threadIdx.x >> 5);
    int m0 = (wave & 3) * 16;   // batch tile
    int h0 = (wave >> 2) * 16;  // hidden tile

    v8f acc[4] = {v8f{}, v8f{}, v8f{}, v8f{}};  // i,f,g,o
    for (int k0 = 0; k0 < Hh; k0 += 32) {
        v16bf a = load_frag_a(h_prev + m0 * Hh + k0, Hh);
        v16bf b[4];
#pragma unroll
        for (int q = 0; q < 4; ++q)
            b[q] = load_frag_b(Whh + (size_t)(q * Hh + h0) * Hh + k0, Hh);
#pragma unroll
        for (int q = 0; q < 4; ++q)
            acc[q] = __builtin_amdgcn_wmma_f32_16x16x32_bf16(
                false, a, false, b[q], (short)0, acc[q], false, false);
    }
    int col = h0 + d_col();
#pragma unroll
    for (int i = 0; i < 8; ++i) {
        int row = m0 + d_row(i);
        const float* xr = xg_t + (size_t)row * Gg;
        float gi = sigmoidf(acc[0][i] + xr[col]);
        float gf = sigmoidf(acc[1][i] + xr[Hh + col]);
        float gg = tanhf  (acc[2][i] + xr[2 * Hh + col]);
        float go = sigmoidf(acc[3][i] + xr[3 * Hh + col]);
        size_t idx = (size_t)row * Hh + col;
        float c = gf * c_state[idx] + gi * gg;
        c_state[idx] = c;
        float h = go * tanhf(c);
        hs_t[idx]   = h;
        h_next[idx] = f32_to_bf16(h);
    }
}

// ---------------------------------------------------------------------------
// Classifier: scores[b,t,c] = hs[t,b,:] . Wc[c,:] + bc[c]  (tiny, VALU fp32)
// ---------------------------------------------------------------------------
__global__ void classifier(const float* __restrict__ hs,   // [T,B,H]
                           const float* __restrict__ Wc,   // [C,H]
                           const float* __restrict__ bc,
                           float* __restrict__ out) {      // [B,T,C]
    int idx = blockIdx.x * blockDim.x + threadIdx.x;
    if (idx >= Bb * Tt * Cc) return;
    int c = idx % Cc;
    int t = (idx / Cc) % Tt;
    int b = idx / (Cc * Tt);
    const float* hp = hs + ((size_t)t * Bb + b) * Hh;
    const float* wp = Wc + (size_t)c * Hh;
    float s = bc[c];
#pragma unroll 4
    for (int k = 0; k < Hh; k += 4) {
        float4 hv = *(const float4*)(hp + k);
        float4 wv = *(const float4*)(wp + k);
        s += hv.x * wv.x + hv.y * wv.y + hv.z * wv.z + hv.w * wv.w;
    }
    out[idx] = s;
}

// ---------------------------------------------------------------------------
// Host launcher
// ---------------------------------------------------------------------------
extern "C" void kernel_launch(void* const* d_in, const int* in_sizes, int n_in,
                              void* d_out, int out_size, void* d_ws, size_t ws_size,
                              hipStream_t stream) {
    const float* x    = (const float*)d_in[0];
    const float* W1   = (const float*)d_in[1];
    const float* b1   = (const float*)d_in[2];
    const float* W_ih = (const float*)d_in[3];
    const float* b_ih = (const float*)d_in[4];
    const float* W_hh = (const float*)d_in[5];
    const float* b_hh = (const float*)d_in[6];
    const float* Wc   = (const float*)d_in[7];
    const float* bc   = (const float*)d_in[8];
    float* out = (float*)d_out;

    // Workspace carve-up (256B aligned)
    char* ws = (char*)d_ws;
    size_t off = 0;
    auto carve = [&](size_t bytes) -> void* {
        off = (off + 255) & ~(size_t)255;
        void* p = ws + off;
        off += bytes;
        return p;
    };
    unsigned short* x_bf   = (unsigned short*)carve((size_t)Mrows * Dd * 2); // 67 MB
    unsigned short* W1_bf  = (unsigned short*)carve((size_t)Pp * Dd * 2);    // 16.8 MB
    unsigned short* Wih_bf = (unsigned short*)carve((size_t)Gg * Pp * 2);    // 8.4 MB
    unsigned short* Whh_bf = (unsigned short*)carve((size_t)Gg * Hh * 2);    // 2.1 MB
    unsigned short* lin_bf = (unsigned short*)carve((size_t)Mrows * Pp * 2); // 33.5 MB
    float*          xg     = (float*)carve((size_t)Tt * Bb * Gg * 4);        // 67 MB
    float*          c_st   = (float*)carve((size_t)Bb * Hh * 4);
    unsigned short* h_buf0 = (unsigned short*)carve((size_t)Bb * Hh * 2);
    unsigned short* h_buf1 = (unsigned short*)carve((size_t)Bb * Hh * 2);
    float*          hs     = (float*)carve((size_t)Tt * Bb * Hh * 4);        // 16.8 MB
    (void)ws_size; (void)n_in; (void)in_sizes; (void)out_size;

    // 1) fp32 -> bf16 conversions
    {
        int n = Mrows * Dd;
        cvt_f32_bf16<<<(n + 255) / 256, 256, 0, stream>>>(x, x_bf, n);
        n = Pp * Dd;
        cvt_f32_bf16<<<(n + 255) / 256, 256, 0, stream>>>(W1, W1_bf, n);
        n = Gg * Pp;
        cvt_f32_bf16<<<(n + 255) / 256, 256, 0, stream>>>(W_ih, Wih_bf, n);
        n = Gg * Hh;
        cvt_f32_bf16<<<(n + 255) / 256, 256, 0, stream>>>(W_hh, Whh_bf, n);
    }

    // 2) zero LSTM state
    {
        int n = Bb * Hh;
        init_state<<<(n + 255) / 256, 256, 0, stream>>>(c_st, h_buf0, n);
    }

    // 3) GEMM1: lin = relu(x @ W1^T + b1)   (4096 waves, 8 waves/block)
    gemm1_relu<<<(Mrows / 64) * (Pp / 64) / 8, 256, 0, stream>>>(x_bf, W1_bf, b1, lin_bf);

    // 4) GEMM2: xg = lin @ W_ih^T + b_ih + b_hh, stored [T,B,4H]
    gemm2_xg<<<(Mrows / 64) * (Gg / 64) / 8, 256, 0, stream>>>(lin_bf, Wih_bf, b_ih, b_hh, xg);

    // 5) LSTM recurrence: one fused kernel per timestep (graph-friendly chain)
    for (int t = 0; t < Tt; ++t) {
        unsigned short* hp = (t & 1) ? h_buf1 : h_buf0;
        unsigned short* hn = (t & 1) ? h_buf0 : h_buf1;
        lstm_step<<<16, 256, 0, stream>>>(hp, Whh_bf,
                                          xg + (size_t)t * Bb * Gg,
                                          c_st, hn,
                                          hs + (size_t)t * Bb * Hh);
    }

    // 6) classifier
    {
        int n = Bb * Tt * Cc;
        classifier<<<(n + 255) / 256, 256, 0, stream>>>(hs, Wc, bc, out);
    }
}
